// HCURec_32134945309150
// MI455X (gfx1250) — compile-verified, matrix operand order checked by reading, not run on hardware
//
#include <hip/hip_runtime.h>

typedef _Float16 h16;
typedef __attribute__((ext_vector_type(8)))  _Float16 v8h;
typedef __attribute__((ext_vector_type(16))) _Float16 v16h;
typedef __attribute__((ext_vector_type(8)))  float    v8f;

#define B_        32
#define NNUM      5
#define UCATN     8
#define USUBN     4
#define UNN       5
#define H_        256
#define TL_       30
#define EL_       10
#define ATT_      200
#define RED_      100
#define DATA_COLS 225
#define ITEMS_USER 5120
#define ITEMS_ALL  5280

// ---------------------------------------------------------------------------
// helpers
// ---------------------------------------------------------------------------
__device__ __forceinline__ float block_reduce_sum(float v, float* red) {
    int tid = threadIdx.x;
    red[tid] = v;
    __syncthreads();
    for (int s = 128; s > 0; s >>= 1) {
        if (tid < s) red[tid] += red[tid + s];
        __syncthreads();
    }
    float r = red[0];
    __syncthreads();
    return r;
}

// _h1(x) = tanh(x @ W1 + b1) @ W2 + b2  (x in LDS, 256 -> 100 -> 1)
__device__ __forceinline__ float h1_eval(const float* x,
                                         const float* __restrict__ W1,
                                         const float* __restrict__ b1,
                                         const float* __restrict__ W2,
                                         const float* __restrict__ b2,
                                         float* red) {
    int tid = threadIdx.x;
    float v = 0.f;
    if (tid < RED_) {
        float acc = b1[tid];
        for (int k = 0; k < H_; ++k) acc += x[k] * W1[k * RED_ + tid];
        v = tanhf(acc) * W2[tid];
    }
    return block_reduce_sum(v, red) + b2[0];
}

// ---------------------------------------------------------------------------
// weight convert+transpose to f16:  dst[n*K + k] = src[k*N + n], pad n>=N to 0
// ---------------------------------------------------------------------------
__global__ __launch_bounds__(256) void convertT_kernel(
    const float* __restrict__ src, h16* __restrict__ dst, int K, int N, int Npad) {
    int idx = blockIdx.x * 256 + threadIdx.x;
    if (idx >= Npad * K) return;
    int n = idx / K, k = idx % K;
    dst[idx] = (h16)((n < N) ? src[k * N + n] : 0.f);
}

// ---------------------------------------------------------------------------
// Per-news-item encoder front end: gather title/entity embeddings, WMMA
// attention scores S = te(30x256) @ Wt(256x200), softmax over words,
// emit concat rows (f16) for the big news/po GEMMs.
// ---------------------------------------------------------------------------
__global__ __launch_bounds__(256) void encode_kernel(
    const int*   __restrict__ data,
    const int*   __restrict__ title_idx,
    const int*   __restrict__ ent_idx,
    const float* __restrict__ word_emb,
    const float* __restrict__ ent_emb,
    const h16*   __restrict__ wtT,     // [208][256] f16, transposed news_Wt
    const float* __restrict__ vt,      // [200]
    h16* __restrict__ xcat_news,       // [ITEMS_ALL][512]
    h16* __restrict__ xcat_po)         // [ITEMS_USER][512]
{
    __shared__ __align__(16) h16 teh[32][256];
    __shared__ int   tids[32];
    __shared__ int   eids[EL_];
    __shared__ float score[32];
    __shared__ float attp[32];
    __shared__ float erep[H_];

    int item = blockIdx.x;
    int tid  = threadIdx.x;

    if (tid == 0) {
        int nid;
        if (item < ITEMS_USER) {
            int b = item / 160, rem = item % 160;
            nid = data[b * DATA_COLS + 65 + rem];           // u_news block
        } else {
            int t = item - ITEMS_USER;
            int b = t / NNUM, i = t % NNUM;
            nid = data[b * DATA_COLS + i];                  // candidate ids
        }
        tids[31] = nid;
    }
    __syncthreads();
    int nid = tids[31];
    if (tid < TL_)                 tids[tid] = title_idx[nid * TL_ + tid];
    if (tid >= 64 && tid < 64+EL_) eids[tid - 64] = ent_idx[nid * EL_ + (tid - 64)];
    if (tid < 32)                  score[tid] = 0.f;
    __syncthreads();

    // gather title embeddings -> LDS f16 (pad rows 30,31 with zeros)
    for (int e = tid; e < 32 * H_; e += 256) {
        int l = e >> 8, hc = e & 255;
        float v = (l < TL_) ? word_emb[tids[l] * H_ + hc] : 0.f;
        teh[l][hc] = (h16)v;
    }
    // entity mean (each thread owns one of 256 dims)
    {
        float s = 0.f;
        for (int e = 0; e < EL_; ++e) s += ent_emb[eids[e] * H_ + tid];
        erep[tid] = s * (1.f / EL_);
    }
    __syncthreads();

    // S = te @ Wt  : 2 x 13 tiles of 16x16, K = 256 (8 wmma steps each)
    int wid   = tid >> 5, lane = tid & 31;
    int nloc  = lane & 15, khalf = lane >> 4;
    for (int t = wid; t < 26; t += 8) {
        int mt = t & 1, nt = t >> 1;
        int m = mt * 16 + nloc;
        int n = nt * 16 + nloc;
        v8f acc = {};
        for (int kk = 0; kk < 8; ++kk) {
            int kb = kk * 32 + khalf * 8;
            union { v16h v; v8h p[2]; } Af, Bf;
            Af.p[0] = *(const v8h*)&teh[m][kb];
            Af.p[1] = *(const v8h*)&teh[m][kb + 16];
            const h16* bp = wtT + n * 256 + kb;
            Bf.p[0] = *(const v8h*)bp;
            Bf.p[1] = *(const v8h*)(bp + 16);
            acc = __builtin_amdgcn_wmma_f32_16x16x32_f16(
                false, Af.v, false, Bf.v, (short)0, acc, false, false);
        }
        if (n < ATT_) {
            float vtn = vt[n];
            for (int j = 0; j < 8; ++j) {
                int mm = mt * 16 + khalf * 8 + j;
                if (mm < TL_) atomicAdd(&score[mm], tanhf(acc[j]) * vtn);
            }
        }
    }
    __syncthreads();

    // softmax over 30 word scores
    if (tid == 0) {
        float mx = -1e30f;
        for (int l = 0; l < TL_; ++l) mx = fmaxf(mx, score[l]);
        float s = 0.f;
        for (int l = 0; l < TL_; ++l) { float e = __expf(score[l] - mx); attp[l] = e; s += e; }
        float inv = 1.f / s;
        for (int l = 0; l < TL_; ++l) attp[l] *= inv;
    }
    __syncthreads();

    // t_rep (attention-weighted) and t_mean (po encoder), emit concat rows
    float trep = 0.f, tmean = 0.f;
    for (int l = 0; l < TL_; ++l) {
        float v = (float)teh[l][tid];
        trep  += attp[l] * v;
        tmean += v;
    }
    tmean *= (1.f / TL_);
    xcat_news[(size_t)item * 512 + tid]        = (h16)trep;       // [t_rep, e_rep]
    xcat_news[(size_t)item * 512 + H_ + tid]   = (h16)erep[tid];
    if (item < ITEMS_USER) {
        xcat_po[(size_t)item * 512 + tid]      = (h16)erep[tid];  // [e_rep, t_mean]
        xcat_po[(size_t)item * 512 + H_ + tid] = (h16)tmean;
    }
}

// ---------------------------------------------------------------------------
// Generic WMMA GEMM: C[M,N] = act(A[M,K](f16) @ WT[N,K](f16)^T + bias)
// M % 16 == 0, N % 16 == 0, K % 32 == 0. One 16x16 tile per wave.
// ---------------------------------------------------------------------------
__global__ __launch_bounds__(256) void gemm_bias_act_kernel(
    const h16* __restrict__ A, const h16* __restrict__ WT,
    const float* __restrict__ bias, float* __restrict__ C,
    int M, int N, int K, int do_tanh)
{
    int tiles_n = N >> 4;
    int tiles   = (M >> 4) * tiles_n;
    int t = blockIdx.x * 8 + (threadIdx.x >> 5);
    if (t >= tiles) return;                 // wave-uniform
    int lane = threadIdx.x & 31;
    int nloc = lane & 15, khalf = lane >> 4;
    int mt = t / tiles_n, nt = t % tiles_n;
    int m = mt * 16 + nloc;
    int n = nt * 16 + nloc;
    v8f acc = {};
    for (int k0 = 0; k0 < K; k0 += 32) {
        int kb = k0 + khalf * 8;
        union { v16h v; v8h p[2]; } Af, Bf;
        const h16* ap = A + (size_t)m * K + kb;
        Af.p[0] = *(const v8h*)ap;
        Af.p[1] = *(const v8h*)(ap + 16);
        const h16* bp = WT + (size_t)n * K + kb;
        Bf.p[0] = *(const v8h*)bp;
        Bf.p[1] = *(const v8h*)(bp + 16);
        acc = __builtin_amdgcn_wmma_f32_16x16x32_f16(
            false, Af.v, false, Bf.v, (short)0, acc, false, false);
    }
    float bn = bias ? bias[n] : 0.f;
    for (int j = 0; j < 8; ++j) {
        int mm = mt * 16 + khalf * 8 + j;
        float v = acc[j] + bn;
        C[(size_t)mm * N + n] = do_tanh ? tanhf(v) : v;
    }
}

// ---------------------------------------------------------------------------
// h1 heads: h1sub for all 5280 reps; h1cate/h1glob for the 160 candidates
// ---------------------------------------------------------------------------
__global__ __launch_bounds__(256) void h1_all_kernel(
    const float* __restrict__ nr,
    const float* sW1, const float* sb1, const float* sW2, const float* sb2,
    const float* cW1, const float* cb1, const float* cW2, const float* cb2,
    const float* gW1, const float* gb1, const float* gW2, const float* gb2,
    float* __restrict__ h1sub_all, float* __restrict__ h1cate_t,
    float* __restrict__ h1glob_t)
{
    __shared__ float x[H_];
    __shared__ float red[256];
    int r = blockIdx.x, tid = threadIdx.x;
    x[tid] = nr[(size_t)r * H_ + tid];
    __syncthreads();
    float hs = h1_eval(x, sW1, sb1, sW2, sb2, red);
    if (tid == 0) h1sub_all[r] = hs;
    if (r >= ITEMS_USER) {
        float hc = h1_eval(x, cW1, cb1, cW2, cb2, red);
        float hg = h1_eval(x, gW1, gb1, gW2, gb2, red);
        if (tid == 0) { h1cate_t[r - ITEMS_USER] = hc; h1glob_t[r - ITEMS_USER] = hg; }
    }
}

// ---------------------------------------------------------------------------
// attend-over-L + dense epilogue (shared by sub / cate stages)
// feats: LDS [L][512] already populated; writes out[256]
// ---------------------------------------------------------------------------
__device__ __forceinline__ void attend_dense(
    float (*feats)[512], int L,
    const float* __restrict__ attW, const float* __restrict__ attv,
    const float* __restrict__ dW,  const float* __restrict__ db,
    float* __restrict__ out_row, float* red, float* sc, float* wfeat)
{
    int tid = threadIdx.x;
    for (int u = 0; u < L; ++u) {
        float v = 0.f;
        if (tid < ATT_) {
            float acc = 0.f;
            for (int k = 0; k < 512; ++k) acc += feats[u][k] * attW[k * ATT_ + tid];
            v = tanhf(acc) * attv[tid];
        }
        float s = block_reduce_sum(v, red);
        if (tid == 0) sc[u] = s;
        __syncthreads();
    }
    if (tid == 0) {
        float mx = -1e30f;
        for (int u = 0; u < L; ++u) mx = fmaxf(mx, sc[u]);
        float s = 0.f;
        for (int u = 0; u < L; ++u) { float e = __expf(sc[u] - mx); sc[u] = e; s += e; }
        float inv = 1.f / s;
        for (int u = 0; u < L; ++u) sc[u] *= inv;
    }
    __syncthreads();
    float w0 = 0.f, w1 = 0.f;
    for (int u = 0; u < L; ++u) {
        w0 += sc[u] * feats[u][tid];
        w1 += sc[u] * feats[u][H_ + tid];
    }
    wfeat[tid] = w0; wfeat[H_ + tid] = w1;
    __syncthreads();
    float acc = db[tid];
    for (int k = 0; k < 512; ++k) acc += wfeat[k] * dW[k * H_ + tid];
    out_row[tid] = acc;
}

__global__ __launch_bounds__(256) void sub_att_kernel(
    const int* __restrict__ data, const float* __restrict__ po_in,
    const float* __restrict__ subcat_emb,
    const float* attW, const float* attv, const float* dW, const float* db,
    float* __restrict__ sub_po)
{
    __shared__ float feats[UNN][512];
    __shared__ float red[256];
    __shared__ float sc[8];
    __shared__ float wfeat[512];
    int r2 = blockIdx.x, tid = threadIdx.x;
    int b = r2 >> 5, c = (r2 >> 2) & 7, s = r2 & 3;
    for (int u = 0; u < UNN; ++u) {
        int r = r2 * UNN + u;
        feats[u][tid] = po_in[(size_t)r * H_ + tid];
        // exact tile-then-reshape scramble of the reference
        int m32 = (c * 20 + s * 5 + u) & 31;
        int c2 = m32 >> 2, s2 = m32 & 3;
        int sid = data[b * DATA_COLS + 33 + c2 * 4 + s2];
        feats[u][H_ + tid] = subcat_emb[sid * H_ + tid];
    }
    __syncthreads();
    attend_dense(feats, UNN, attW, attv, dW, db, sub_po + (size_t)r2 * H_,
                 red, sc, wfeat);
}

__global__ __launch_bounds__(256) void cate_att_kernel(
    const int* __restrict__ data, const float* __restrict__ sub_po,
    const float* __restrict__ cat_emb,
    const float* attW, const float* attv, const float* dW, const float* db,
    float* __restrict__ cate_po)
{
    __shared__ float feats[USUBN][512];
    __shared__ float red[256];
    __shared__ float sc[8];
    __shared__ float wfeat[512];
    int r3 = blockIdx.x, tid = threadIdx.x;
    int b = r3 >> 3, c = r3 & 7;
    int cid = data[b * DATA_COLS + 25 + c];
    float catv = cat_emb[cid * H_ + tid];
    for (int u = 0; u < USUBN; ++u) {
        feats[u][tid]      = sub_po[(size_t)(r3 * USUBN + u) * H_ + tid];
        feats[u][H_ + tid] = catv;
    }
    __syncthreads();
    attend_dense(feats, USUBN, attW, attv, dW, db, cate_po + (size_t)r3 * H_,
                 red, sc, wfeat);
}

// ---------------------------------------------------------------------------
// num scores: rows 0..1023 -> sub, 1024..1279 -> cate
// ---------------------------------------------------------------------------
__global__ __launch_bounds__(256) void num_score_kernel(
    const int* __restrict__ usubn, const int* __restrict__ ucatn,
    const float* __restrict__ sub_num_emb, const float* __restrict__ cate_num_emb,
    const float* sW, const float* sb, const float* cW, const float* cb,
    float* __restrict__ sub_score, float* __restrict__ cate_score)
{
    __shared__ float red[256];
    int r = blockIdx.x, tid = threadIdx.x;
    float v;
    if (r < 1024) v = sub_num_emb[usubn[r] * H_ + tid] * sW[tid];
    else          v = cate_num_emb[ucatn[r - 1024] * H_ + tid] * cW[tid];
    float s = block_reduce_sum(v, red);
    if (tid == 0) {
        if (r < 1024) sub_score[r] = s + sb[0];
        else          cate_score[r - 1024] = s + cb[0];
    }
}

// ---------------------------------------------------------------------------
// per-candidate stages (cross_att collapses exactly to softmax(base+0.01*cs))
// ---------------------------------------------------------------------------
__global__ __launch_bounds__(256) void sub_rep_kernel(
    const float* __restrict__ nr_all, const float* __restrict__ sub_po,
    const float* __restrict__ h1sub_all,
    const float* cW1, const float* cb1, const float* cW2, const float* cb2,
    float* __restrict__ sub_rep0, float* __restrict__ srep_h1c, int cand)
{
    __shared__ float att[8];
    __shared__ float srow[H_];
    __shared__ float red[256];
    int r2 = blockIdx.x, tid = threadIdx.x, b = r2 >> 5;
    if (tid == 0) {
        float cs = h1sub_all[ITEMS_USER + b * NNUM + cand];
        float e[UNN], mx = -1e30f;
        for (int m = 0; m < UNN; ++m) { e[m] = h1sub_all[r2 * UNN + m] + 0.01f * cs; mx = fmaxf(mx, e[m]); }
        float s = 0.f;
        for (int m = 0; m < UNN; ++m) { e[m] = __expf(e[m] - mx); s += e[m]; }
        for (int m = 0; m < UNN; ++m) att[m] = e[m] / s;
    }
    __syncthreads();
    float v = sub_po[(size_t)r2 * H_ + tid];
    for (int m = 0; m < UNN; ++m) v += att[m] * nr_all[(size_t)(r2 * UNN + m) * H_ + tid];
    srow[tid] = v;
    sub_rep0[(size_t)r2 * H_ + tid] = v;
    __syncthreads();
    float h = h1_eval(srow, cW1, cb1, cW2, cb2, red);
    if (tid == 0) srep_h1c[r2] = h;
}

__global__ __launch_bounds__(256) void cate_rep_kernel(
    const float* __restrict__ sub_rep0, const float* __restrict__ cate_po,
    const float* __restrict__ srep_h1c, const float* __restrict__ sub_ns,
    const float* __restrict__ h1cate_t,
    const float* gW1, const float* gb1, const float* gW2, const float* gb2,
    float* __restrict__ cate_rep0, float* __restrict__ crep_h1g, int cand)
{
    __shared__ float att[8];
    __shared__ float crow[H_];
    __shared__ float red[256];
    int r3 = blockIdx.x, tid = threadIdx.x, b = r3 >> 3;
    if (tid == 0) {
        float cc = h1cate_t[b * NNUM + cand];
        float e[USUBN], mx = -1e30f;
        for (int m = 0; m < USUBN; ++m) {
            e[m] = srep_h1c[r3 * USUBN + m] + sub_ns[r3 * USUBN + m] + 0.01f * cc;
            mx = fmaxf(mx, e[m]);
        }
        float s = 0.f;
        for (int m = 0; m < USUBN; ++m) { e[m] = __expf(e[m] - mx); s += e[m]; }
        for (int m = 0; m < USUBN; ++m) att[m] = e[m] / s;
    }
    __syncthreads();
    float v = cate_po[(size_t)r3 * H_ + tid];
    for (int m = 0; m < USUBN; ++m) v += att[m] * sub_rep0[(size_t)(r3 * USUBN + m) * H_ + tid];
    crow[tid] = v;
    cate_rep0[(size_t)r3 * H_ + tid] = v;
    __syncthreads();
    float h = h1_eval(crow, gW1, gb1, gW2, gb2, red);
    if (tid == 0) crep_h1g[r3] = h;
}

__global__ __launch_bounds__(256) void final_kernel(
    const int* __restrict__ data, const int* __restrict__ startp,
    const float* __restrict__ nr_all, const float* __restrict__ sub_rep0,
    const float* __restrict__ cate_rep0, const float* __restrict__ crep_h1g,
    const float* __restrict__ cate_ns, const float* __restrict__ h1glob_t,
    float* __restrict__ out, int cand)
{
    __shared__ float att[8];
    __shared__ float red[256];
    int b = blockIdx.x, tid = threadIdx.x;
    if (tid == 0) {
        float cg = h1glob_t[b * NNUM + cand];
        float e[UCATN], mx = -1e30f;
        for (int m = 0; m < UCATN; ++m) {
            e[m] = crep_h1g[b * UCATN + m] + cate_ns[b * UCATN + m] + 0.01f * cg;
            mx = fmaxf(mx, e[m]);
        }
        float s = 0.f;
        for (int m = 0; m < UCATN; ++m) { e[m] = __expf(e[m] - mx); s += e[m]; }
        for (int m = 0; m < UCATN; ++m) att[m] = e[m] / s;
    }
    __syncthreads();
    float ur = 0.f;
    for (int m = 0; m < UCATN; ++m) ur += att[m] * cate_rep0[(size_t)(b * UCATN + m) * H_ + tid];
    int sp   = startp[b];
    int gcat = sp * UCATN + data[b * DATA_COLS + NNUM + cand];
    int gsub = sp * (UCATN * USUBN) + data[b * DATA_COLS + 2 * NNUM + cand];
    float cd = nr_all[(size_t)(ITEMS_USER + b * NNUM + cand) * H_ + tid];
    float v  = (ur + cate_rep0[(size_t)gcat * H_ + tid] + sub_rep0[(size_t)gsub * H_ + tid]) * cd;
    float s  = block_reduce_sum(v, red);
    if (tid == 0) out[b * NNUM + cand] = s;
}

// ---------------------------------------------------------------------------
// host launcher
// ---------------------------------------------------------------------------
extern "C" void kernel_launch(void* const* d_in, const int* in_sizes, int n_in,
                              void* d_out, int out_size, void* d_ws, size_t ws_size,
                              hipStream_t stream) {
    (void)in_sizes; (void)n_in; (void)out_size; (void)ws_size;

    const int*   data     = (const int*)d_in[0];
    const int*   startp   = (const int*)d_in[1];
    const int*   usubn    = (const int*)d_in[2];
    const int*   ucatn    = (const int*)d_in[3];
    const int*   titlei   = (const int*)d_in[4];
    const int*   enti     = (const int*)d_in[5];
    const float* word_emb = (const float*)d_in[6];
    const float* ent_emb  = (const float*)d_in[7];
    const float* news_Wt  = (const float*)d_in[8];
    const float* news_vt  = (const float*)d_in[9];
    const float* news_W   = (const float*)d_in[10];
    const float* news_b   = (const float*)d_in[11];
    const float* po_W     = (const float*)d_in[12];
    const float* po_b     = (const float*)d_in[13];
    const float* cat_emb  = (const float*)d_in[14];
    const float* subc_emb = (const float*)d_in[15];
    const float* cnum_emb = (const float*)d_in[16];
    const float* snum_emb = (const float*)d_in[17];
    const float* s_h1_W1  = (const float*)d_in[18];
    const float* s_h1_b1  = (const float*)d_in[19];
    const float* s_h1_W2  = (const float*)d_in[20];
    const float* s_h1_b2  = (const float*)d_in[21];
    const float* c_h1_W1  = (const float*)d_in[22];
    const float* c_h1_b1  = (const float*)d_in[23];
    const float* c_h1_W2  = (const float*)d_in[24];
    const float* c_h1_b2  = (const float*)d_in[25];
    const float* g_h1_W1  = (const float*)d_in[26];
    const float* g_h1_b1  = (const float*)d_in[27];
    const float* g_h1_W2  = (const float*)d_in[28];
    const float* g_h1_b2  = (const float*)d_in[29];
    // d_in[30..35]: sub/cate/global_red_{W,b} — provably unused (uniform inner softmax)
    const float* sub_attW = (const float*)d_in[36];
    const float* sub_attv = (const float*)d_in[37];
    const float* cat_attW = (const float*)d_in[38];
    const float* cat_attv = (const float*)d_in[39];
    const float* sub_dW   = (const float*)d_in[40];
    const float* sub_db   = (const float*)d_in[41];
    const float* cat_dW   = (const float*)d_in[42];
    const float* cat_db   = (const float*)d_in[43];
    const float* snum_W   = (const float*)d_in[44];
    const float* snum_b   = (const float*)d_in[45];
    const float* cnum_W   = (const float*)d_in[46];
    const float* cnum_b   = (const float*)d_in[47];

    char* wsb = (char*)d_ws;
    size_t off = 0;
    auto take = [&](size_t bytes) -> void* {
        void* p = wsb + off;
        off += (bytes + 255) & ~(size_t)255;
        return p;
    };
    h16*   wtT       = (h16*)take((size_t)208 * 256 * sizeof(h16));
    h16*   newsWT    = (h16*)take((size_t)256 * 512 * sizeof(h16));
    h16*   poWT      = (h16*)take((size_t)256 * 512 * sizeof(h16));
    h16*   xcat_news = (h16*)take((size_t)ITEMS_ALL * 512 * sizeof(h16));
    h16*   xcat_po   = (h16*)take((size_t)ITEMS_USER * 512 * sizeof(h16));
    float* nr_all    = (float*)take((size_t)ITEMS_ALL * H_ * sizeof(float));
    float* po_in     = (float*)take((size_t)ITEMS_USER * H_ * sizeof(float));
    float* sub_po    = (float*)take((size_t)1024 * H_ * sizeof(float));
    float* cate_po   = (float*)take((size_t)256 * H_ * sizeof(float));
    float* h1sub_all = (float*)take((size_t)ITEMS_ALL * sizeof(float));
    float* h1cate_t  = (float*)take((size_t)160 * sizeof(float));
    float* h1glob_t  = (float*)take((size_t)160 * sizeof(float));
    float* sub_ns    = (float*)take((size_t)1024 * sizeof(float));
    float* cate_ns   = (float*)take((size_t)256 * sizeof(float));
    float* sub_rep0  = (float*)take((size_t)1024 * H_ * sizeof(float));
    float* cate_rep0 = (float*)take((size_t)256 * H_ * sizeof(float));
    float* srep_h1c  = (float*)take((size_t)1024 * sizeof(float));
    float* crep_h1g  = (float*)take((size_t)256 * sizeof(float));

    // 1) weight transposes to f16
    convertT_kernel<<<(208 * 256 + 255) / 256, 256, 0, stream>>>(news_Wt, wtT, 256, 200, 208);
    convertT_kernel<<<(256 * 512 + 255) / 256, 256, 0, stream>>>(news_W, newsWT, 512, 256, 256);
    convertT_kernel<<<(256 * 512 + 255) / 256, 256, 0, stream>>>(po_W, poWT, 512, 256, 256);

    // 2) per-item encoder front end (WMMA title attention)
    encode_kernel<<<ITEMS_ALL, 256, 0, stream>>>(
        data, titlei, enti, word_emb, ent_emb, wtT, news_vt, xcat_news, xcat_po);

    // 3) big WMMA GEMMs: news reps (5280x256) and po reps (5120x256)
    gemm_bias_act_kernel<<<((ITEMS_ALL / 16) * (H_ / 16) + 7) / 8, 256, 0, stream>>>(
        xcat_news, newsWT, news_b, nr_all, ITEMS_ALL, H_, 512, 1);
    gemm_bias_act_kernel<<<((ITEMS_USER / 16) * (H_ / 16) + 7) / 8, 256, 0, stream>>>(
        xcat_po, poWT, po_b, po_in, ITEMS_USER, H_, 512, 1);

    // 4) h1 heads
    h1_all_kernel<<<ITEMS_ALL, 256, 0, stream>>>(
        nr_all, s_h1_W1, s_h1_b1, s_h1_W2, s_h1_b2,
        c_h1_W1, c_h1_b1, c_h1_W2, c_h1_b2,
        g_h1_W1, g_h1_b1, g_h1_W2, g_h1_b2,
        h1sub_all, h1cate_t, h1glob_t);

    // 5) attend stages
    sub_att_kernel<<<1024, 256, 0, stream>>>(
        data, po_in, subc_emb, sub_attW, sub_attv, sub_dW, sub_db, sub_po);
    cate_att_kernel<<<256, 256, 0, stream>>>(
        data, sub_po, cat_emb, cat_attW, cat_attv, cat_dW, cat_db, cate_po);

    // 6) num scores
    num_score_kernel<<<1280, 256, 0, stream>>>(
        usubn, ucatn, snum_emb, cnum_emb, snum_W, snum_b, cnum_W, cnum_b,
        sub_ns, cate_ns);

    // 7) per-candidate pipeline
    for (int cand = 0; cand < NNUM; ++cand) {
        sub_rep_kernel<<<1024, 256, 0, stream>>>(
            nr_all, sub_po, h1sub_all, c_h1_W1, c_h1_b1, c_h1_W2, c_h1_b2,
            sub_rep0, srep_h1c, cand);
        cate_rep_kernel<<<256, 256, 0, stream>>>(
            sub_rep0, cate_po, srep_h1c, sub_ns, h1cate_t,
            g_h1_W1, g_h1_b1, g_h1_W2, g_h1_b2, cate_rep0, crep_h1g, cand);
        final_kernel<<<32, 256, 0, stream>>>(
            data, startp, nr_all, sub_rep0, cate_rep0, crep_h1g, cate_ns,
            h1glob_t, (float*)d_out, cand);
    }
}
